// Discriminator_32014686224952
// MI455X (gfx1250) — compile-verified, matrix operand order checked by reading, not run on hardware
//
#include <hip/hip_runtime.h>
#include <stdint.h>

#define NNODES 100000
#define KDIM   128

typedef __attribute__((ext_vector_type(16))) __bf16 v16bf;
typedef __attribute__((ext_vector_type(8)))  __bf16 v8bf;
typedef __attribute__((ext_vector_type(8)))  float  v8f;

__device__ __forceinline__ __bf16 f2bf(float f) {
  unsigned u = __builtin_bit_cast(unsigned, f);
  unsigned r = u + 0x7FFFu + ((u >> 16) & 1u);   // round-to-nearest-even
  unsigned short s = (unsigned short)(r >> 16);
  return __builtin_bit_cast(__bf16, s);
}

__device__ __forceinline__ void atomic_add_f32(float* p, float v) {
  __hip_atomic_fetch_add(p, v, __ATOMIC_RELAXED, __HIP_MEMORY_SCOPE_AGENT);
}

// ---------------------------------------------------------------- utilities
__global__ void zero_f32(float* __restrict__ p, long long n) {
  long long i = blockIdx.x * (long long)blockDim.x + threadIdx.x;
  long long stride = (long long)gridDim.x * blockDim.x;
  for (; i < n; i += stride) p[i] = 0.0f;
}

__global__ void degree_kernel(const int* __restrict__ src, const int* __restrict__ dst,
                              float* __restrict__ outdeg, float* __restrict__ indeg, int E) {
  int e = blockIdx.x * blockDim.x + threadIdx.x;
  if (e < E) {
    atomic_add_f32(&outdeg[src[e]], 1.0f);
    atomic_add_f32(&indeg[dst[e]], 1.0f);
  }
}

__global__ void norm_kernel(const float* __restrict__ outdeg, const float* __restrict__ indeg,
                            float* __restrict__ nsrc, float* __restrict__ ndst, int n) {
  int i = blockIdx.x * blockDim.x + threadIdx.x;
  if (i < n) {
    nsrc[i] = rsqrtf(fmaxf(outdeg[i], 1.0f));
    ndst[i] = rsqrtf(fmaxf(indeg[i], 1.0f));
  }
}

// ---------------------------------------------------------------- dense GEMM (WMMA bf16)
// h[n, :] = (x[n, :] * nsrc[n]) @ W    W: [KDIM x MOUT] row-major
template <int MOUT>
__global__ __launch_bounds__(256)
void gcn_gemm(const float* __restrict__ x, const float* __restrict__ nsrc,
              const float* __restrict__ W, float* __restrict__ h, int nrows) {
  constexpr int LDK = KDIM + 8;               // pad: 272B row stride, 16B aligned
  __shared__ __bf16 As[64 * LDK];             // 64 rows of norm-scaled activations
  __shared__ __bf16 Bs[MOUT * LDK];           // W transposed: [n][k]

  const int tid  = threadIdx.x;
  const int row0 = blockIdx.x * 64;

  // stage A (coalesced global read, row-major LDS write)
  for (int idx = tid; idx < 64 * KDIM; idx += 256) {
    int r = idx >> 7, k = idx & (KDIM - 1);
    int row = row0 + r;
    float v = 0.0f;
    if (row < nrows) v = x[(size_t)row * KDIM + k] * nsrc[row];
    As[r * LDK + k] = f2bf(v);
  }
  // stage W transposed (coalesced global read along n)
  for (int idx = tid; idx < MOUT * KDIM; idx += 256) {
    int k = idx / MOUT, n = idx % MOUT;
    Bs[n * LDK + k] = f2bf(W[(size_t)k * MOUT + n]);
  }
  __syncthreads();

  const int wave = tid >> 5, lane = tid & 31;
  const int half = lane >> 4, mn = lane & 15;
  constexpr int CT = MOUT / 16;               // col tiles
  constexpr int NT = 4 * CT;                  // 4 row tiles x CT col tiles
  const bool full = (row0 + 64) <= nrows;     // block-uniform tail test

  // B fragments are tile-loop invariant per wave (tc = wave % CT): hoist once.
  const int tc = wave % CT;
  const __bf16* brow = &Bs[(tc * 16 + mn) * LDK];
  v16bf bfrag[KDIM / 32];
#pragma unroll
  for (int kk = 0; kk < KDIM / 32; ++kk) {
    int k0 = kk * 32 + half * 8;              // ISA 16-bit fragment layout
    v8bf bl = *(const v8bf*)(brow + k0);
    v8bf bh = *(const v8bf*)(brow + k0 + 16);
#pragma unroll
    for (int i = 0; i < 8; ++i) { bfrag[kk][i] = bl[i]; bfrag[kk][i + 8] = bh[i]; }
  }

#pragma unroll
  for (int it = 0; it < NT / 8; ++it) {       // compile-time trip count
    const int t  = wave + it * 8;
    const int tr = t / CT;
    const __bf16* arow = &As[(tr * 16 + mn) * LDK];

    // load ALL A k-chunks first: 8x ds_load_b128, one wait, then WMMA chain
    v16bf afrag[KDIM / 32];
#pragma unroll
    for (int kk = 0; kk < KDIM / 32; ++kk) {
      int k0 = kk * 32 + half * 8;
      v8bf al = *(const v8bf*)(arow + k0);
      v8bf ah = *(const v8bf*)(arow + k0 + 16);
#pragma unroll
      for (int i = 0; i < 8; ++i) { afrag[kk][i] = al[i]; afrag[kk][i + 8] = ah[i]; }
    }

    v8f c = {};
#pragma unroll
    for (int kk = 0; kk < KDIM / 32; ++kk)
      c = __builtin_amdgcn_wmma_f32_16x16x32_bf16(false, afrag[kk], false, bfrag[kk],
                                                  (short)0, c, false, false);

    // D layout: VGPR r -> m = 8*half + r, n = lane&15
    const int gr0 = row0 + tr * 16 + half * 8;
    const int gc  = tc * 16 + mn;
    float* out = h + (size_t)gr0 * MOUT + gc;
    if (full) {                               // branch-free bulk path (all but last block)
#pragma unroll
      for (int r = 0; r < 8; ++r) out[(size_t)r * MOUT] = c[r];
    } else {                                  // tail block only
#pragma unroll
      for (int r = 0; r < 8; ++r)
        if (gr0 + r < nrows) out[(size_t)r * MOUT] = c[r];
    }
  }
}

// ---------------------------------------------------------------- edge scatter (L2-resident atomics)
// agg[dst[e], :] += h[src[e], :]   one thread = one (edge, float4 group)
__global__ void scatter_edges(const float4* __restrict__ h4, const int* __restrict__ src,
                              const int* __restrict__ dst, float* __restrict__ agg,
                              int E, int gshift) {
  long long gid = blockIdx.x * (long long)blockDim.x + threadIdx.x;
  int groups = 1 << gshift;
  int g = (int)(gid & (groups - 1));
  long long e = gid >> gshift;
  if (e >= E) return;
  int s = src[e], d = dst[e];
  float4 v = h4[(size_t)s * groups + g];
  float* base = agg + ((size_t)d << (gshift + 2)) + g * 4;
  atomic_add_f32(base + 0, v.x);
  atomic_add_f32(base + 1, v.y);
  atomic_add_f32(base + 2, v.z);
  atomic_add_f32(base + 3, v.w);
}

// ---------------------------------------------------------------- epilogue
template <int F, bool RELU>
__global__ void finish_kernel(const float* __restrict__ agg, const float* __restrict__ ndst,
                              const float* __restrict__ bias, float* __restrict__ out, int n) {
  long long idx = blockIdx.x * (long long)blockDim.x + threadIdx.x;
  if (idx >= (long long)n * F) return;
  int node = (int)(idx / F);
  int f = (int)(idx & (F - 1));
  float v = agg[idx] * ndst[node] + bias[f];
  out[idx] = RELU ? fmaxf(v, 0.0f) : v;
}

// ---------------------------------------------------------------- driver
extern "C" void kernel_launch(void* const* d_in, const int* in_sizes, int n_in,
                              void* d_out, int out_size, void* d_ws, size_t ws_size,
                              hipStream_t stream) {
  const float* in_feat = (const float*)d_in[0];
  const int*   src     = (const int*)d_in[1];
  const int*   dst     = (const int*)d_in[2];
  const float* W1 = (const float*)d_in[3];  const float* b1 = (const float*)d_in[4];
  const float* W2 = (const float*)d_in[5];  const float* b2 = (const float*)d_in[6];
  const float* W3 = (const float*)d_in[7];  const float* b3 = (const float*)d_in[8];
  const float* W4 = (const float*)d_in[9];  const float* b4 = (const float*)d_in[10];

  const int N = NNODES;
  const int E = in_sizes[1];

  float* ws     = (float*)d_ws;
  float* outdeg = ws;
  float* indeg  = outdeg + N;
  float* nsrc   = indeg + N;
  float* ndst   = nsrc + N;
  float* bufA   = ndst + N;                    // activation ping
  float* bufB   = bufA + (size_t)N * 128;      // gemm output h
  float* bufC   = bufB + (size_t)N * 128;      // aggregation target

  const dim3 blk(256);
  const unsigned gemm_blocks = (N + 63) / 64;
  const unsigned nf128 = (unsigned)(((long long)N * 128 + 255) / 256);
  const unsigned nf64  = (unsigned)(((long long)N * 64 + 255) / 256);
  const unsigned sc128 = (unsigned)(((long long)E * 32 + 255) / 256);
  const unsigned sc64  = (unsigned)(((long long)E * 16 + 255) / 256);

  // degree norms
  zero_f32<<<2048, blk, 0, stream>>>(outdeg, 2LL * N);
  degree_kernel<<<(E + 255) / 256, blk, 0, stream>>>(src, dst, outdeg, indeg, E);
  norm_kernel<<<(N + 255) / 256, blk, 0, stream>>>(outdeg, indeg, nsrc, ndst, N);

  // layer 1
  gcn_gemm<128><<<gemm_blocks, blk, 0, stream>>>(in_feat, nsrc, W1, bufB, N);
  zero_f32<<<4096, blk, 0, stream>>>(bufC, (long long)N * 128);
  scatter_edges<<<sc128, blk, 0, stream>>>((const float4*)bufB, src, dst, bufC, E, 5);
  finish_kernel<128, true><<<nf128, blk, 0, stream>>>(bufC, ndst, b1, bufA, N);

  // layer 2
  gcn_gemm<128><<<gemm_blocks, blk, 0, stream>>>(bufA, nsrc, W2, bufB, N);
  zero_f32<<<4096, blk, 0, stream>>>(bufC, (long long)N * 128);
  scatter_edges<<<sc128, blk, 0, stream>>>((const float4*)bufB, src, dst, bufC, E, 5);
  finish_kernel<128, true><<<nf128, blk, 0, stream>>>(bufC, ndst, b2, bufA, N);

  // layer 3
  gcn_gemm<128><<<gemm_blocks, blk, 0, stream>>>(bufA, nsrc, W3, bufB, N);
  zero_f32<<<4096, blk, 0, stream>>>(bufC, (long long)N * 128);
  scatter_edges<<<sc128, blk, 0, stream>>>((const float4*)bufB, src, dst, bufC, E, 5);
  finish_kernel<128, true><<<nf128, blk, 0, stream>>>(bufC, ndst, b3, bufA, N);

  // layer 4 (128 -> 64, no relu, write d_out)
  gcn_gemm<64><<<gemm_blocks, blk, 0, stream>>>(bufA, nsrc, W4, bufB, N);
  zero_f32<<<4096, blk, 0, stream>>>(bufC, (long long)N * 64);
  scatter_edges<<<sc64, blk, 0, stream>>>((const float4*)bufB, src, dst, bufC, E, 4);
  finish_kernel<64, false><<<nf64, blk, 0, stream>>>(bufC, ndst, b4, (float*)d_out, N);
}
